// EquiformerLayer_71992241815599
// MI455X (gfx1250) — matrix operand order repaired and checked by reading, not compiled, not run on hardware
//
#include <hip/hip_runtime.h>
#include <stdint.h>

#define SQ3     1.7320508075688772f
#define INV_SQ3 0.5773502691896258f
#define INV_MUL 0.125f                  /* 1/sqrt(64)  */
#define INV_TP  0.08838834764831845f    /* 1/sqrt(128) */

typedef __bf16 bf16_t;
typedef bf16_t v16bf __attribute__((ext_vector_type(16)));
typedef float  v8f   __attribute__((ext_vector_type(8)));
typedef unsigned int uv4 __attribute__((ext_vector_type(4)));
typedef int          iv8 __attribute__((ext_vector_type(8)));
typedef int          iv4 __attribute__((ext_vector_type(4)));

// weight slots (13 used matrices; lin_dst is dead in the reference forward)
#define SLOT_LS0   0
#define SLOT_LS1   1
#define SLOT_T1W00 2
#define SLOT_T1W11 3
#define SLOT_T1W01 4
#define SLOT_T1W10 5
#define SLOT_T2W00 6
#define SLOT_T2W11 7
#define SLOT_T2W01 8
#define SLOT_T2W10 9
#define SLOT_LH0   10
#define SLOT_LH1   11
#define SLOT_LSC   12
#define NSLOTS     13
#define WWORDS     (NSLOTS * 2048)      /* 26624 dwords = 104 KB */

union FragU { uint32_t w[8]; uint4 q[2]; v16bf v; };

__device__ __forceinline__ uint16_t f2bf(float f) {
    uint32_t u = __float_as_uint(f);
    uint32_t r = ((u >> 16) & 1u) + 0x7fffu;   // round-to-nearest-even
    return (uint16_t)((u + r) >> 16);
}
__device__ __forceinline__ float bf2f(uint16_t h) {
    return __uint_as_float(((uint32_t)h) << 16);
}

__device__ __forceinline__ v8f wmma_bf(v16bf a, v16bf b, v8f c) {
    return __builtin_amdgcn_wmma_f32_16x16x32_bf16(false, a, false, b, (short)0, c, false, false);
}

// A-fragment: 16x32 bf16 from row-major bf16[16][64] tile in LDS.
// ISA layout: lanes0-15 M=0..15 {V0..3:K0..7, V4..7:K16..23}; lanes16-31 {K8..15, K24..31}.
// Per lane: two contiguous 4-word runs -> two ds_load_b128.
__device__ __forceinline__ v16bf load_a(const uint16_t* tile, int kb, int lane) {
    const uint32_t* t = (const uint32_t*)tile;
    int base = (lane & 15) * 32 + kb * 16 + (lane >> 4) * 4;
    FragU f;
    f.q[0] = *(const uint4*)(t + base);
    f.q[1] = *(const uint4*)(t + base + 8);
    return f.v;
}

// B-fragment: weights pre-swizzled fragment-major; each lane's 8 words contiguous.
__device__ __forceinline__ v16bf load_b(const uint32_t* wlds, int slot, int kb, int nb, int lane) {
    int base = ((((slot * 2 + kb) * 4 + nb) * 32) + lane) * 8;
    FragU f;
    f.q[0] = *(const uint4*)(wlds + base);
    f.q[1] = *(const uint4*)(wlds + base + 4);
    return f.v;
}

// C[16x64] = T[16x64] @ W[64x64]
__device__ __forceinline__ void mm1(v8f c[4], const uint16_t* t, const uint32_t* wlds,
                                    int slot, int lane) {
    v16bf a0 = load_a(t, 0, lane), a1 = load_a(t, 1, lane);
#pragma unroll
    for (int nb = 0; nb < 4; ++nb) {
        v8f acc = {0.f, 0.f, 0.f, 0.f, 0.f, 0.f, 0.f, 0.f};
        acc = wmma_bf(a0, load_b(wlds, slot, 0, nb, lane), acc);
        acc = wmma_bf(a1, load_b(wlds, slot, 1, nb, lane), acc);
        c[nb] = acc;
    }
}

// C = T1@W1 + T2@W2
__device__ __forceinline__ void mm2(v8f c[4], const uint16_t* t1, int slot1,
                                    const uint16_t* t2, int slot2,
                                    const uint32_t* wlds, int lane) {
    v16bf a0 = load_a(t1, 0, lane), a1 = load_a(t1, 1, lane);
    v16bf d0 = load_a(t2, 0, lane), d1 = load_a(t2, 1, lane);
#pragma unroll
    for (int nb = 0; nb < 4; ++nb) {
        v8f acc = {0.f, 0.f, 0.f, 0.f, 0.f, 0.f, 0.f, 0.f};
        acc = wmma_bf(a0, load_b(wlds, slot1, 0, nb, lane), acc);
        acc = wmma_bf(a1, load_b(wlds, slot1, 1, nb, lane), acc);
        acc = wmma_bf(d0, load_b(wlds, slot2, 0, nb, lane), acc);
        acc = wmma_bf(d1, load_b(wlds, slot2, 1, nb, lane), acc);
        c[nb] = acc;
    }
}

__device__ __forceinline__ void atomic_add_f32(float* p, float v) {
    unsafeAtomicAdd(p, v);   // -> global_atomic_add_f32
}

struct WaveLds {
    uint16_t xs[4][1024];   // x0,x1x,x1y,x1z  -> later h0,h1x,h1y,h1z (post fctp1)
    uint16_t s[4][1024];    // s0,s1x,s1y,s1z  -> later h0',h1m'      (post lin_hidden)
    uint16_t dott[1024];    // dot (fctp1), then dot2 (fctp2)
    uint16_t h0r[1024];     // leaky_relu(h0')
    float    act[1024];     // sc then softmax activations (scalar channels)
    float    sh[16][4];     // sqrt3 * normalized edge vector
    float    a1[16];        // softmax prob of a structurally-zero channel
    int      srcv[16];
    int      dstv[16];
};

// ---- pack weights fragment-major bf16: wp[slot][kb][nb][lane][j] ----
__global__ void pack_weights_kernel(
    const float* w0, const float* w1, const float* w2, const float* w3,
    const float* w4, const float* w5, const float* w6, const float* w7,
    const float* w8, const float* w9, const float* w10, const float* w11,
    const float* w12, const float* w13, const float* w14, uint32_t* wp)
{
    const float* ws[NSLOTS] = {w0, w1, w4, w5, w6, w7, w8, w9, w10, w11, w12, w13, w14};
    int t = blockIdx.x * 256 + threadIdx.x;
    if (t >= WWORDS) return;
    int slot = t >> 11, r = t & 2047;
    int j = r & 7, lane = (r >> 3) & 31, nb = (r >> 8) & 3, kb = r >> 10;
    int kp = (j & 3) + 4 * (lane >> 4) + 8 * (j >> 2);  // B-layout: lanes0-15 K=0..7,16..23; lanes16-31 K=8..15,24..31
    int n  = nb * 16 + (lane & 15);
    int k0 = kb * 32 + 2 * kp;
    const float* W = ws[slot];
    uint32_t lo = f2bf(W[k0 * 64 + n]);
    uint32_t hi = f2bf(W[(k0 + 1) * 64 + n]);
    wp[t] = lo | (hi << 16);
}

__global__ void init_out_kernel(const float* __restrict__ atom, float* __restrict__ out, int total) {
    int t = blockIdx.x * 256 + threadIdx.x;
    if (t < total) out[t] = atom[t];
}

__global__ void __launch_bounds__(128)
equiformer_edge_kernel(const float* __restrict__ atom,
                       const float* __restrict__ evec,
                       const int*   __restrict__ eidx,
                       const uint32_t* __restrict__ wp,
                       float* __restrict__ out, int E)
{
    __shared__ uint32_t wlds[WWORDS];   // 104 KB packed weights, block-resident
    __shared__ WaveLds  smem[4];        // 4 waves x 25 KB staging
    const int wave = threadIdx.x >> 5;
    const int lane = threadIdx.x & 31;
    const int ln = lane & 15;
    const int g  = lane >> 4;
    WaveLds& L = smem[wave];

    // ---- stage all weights into LDS once per block via the Tensor Data Mover ----
#if __has_builtin(__builtin_amdgcn_tensor_load_to_lds)
    if (wave == 0) {
        uint64_t ga = (uint64_t)(uintptr_t)wp;
        uint32_t lo32 = (uint32_t)(uintptr_t)(void*)&wlds[0];  // flat low 32 = LDS offset
        uv4 g0;
        g0[0] = 1u;                                            // count=1, user descriptor
        g0[1] = lo32;                                          // lds_addr
        g0[2] = (uint32_t)ga;                                  // global_addr[31:0]
        g0[3] = (uint32_t)((ga >> 32) & 0x1FFFFFFu) | (2u << 30);  // addr[56:32], type=2
        iv8 g1;
        g1[0] = (int)(2u << 16);                               // data_size=4B, wg_mask=0
        g1[1] = (int)(((uint32_t)WWORDS & 0xFFFFu) << 16);     // tensor_dim0[15:0]
        g1[2] = (int)(((uint32_t)WWORDS >> 16) | (1u << 16));  // tensor_dim0[31:16], tensor_dim1=1
        g1[3] = (int)(((uint32_t)WWORDS & 0xFFFFu) << 16);     // tile_dim0 = WWORDS
        g1[4] = 0;                                             // tile_dim1/2 unused
        g1[5] = (int)WWORDS;                                   // tensor_dim0_stride
        g1[6] = 0; g1[7] = 0;
        iv4 z4 = {0, 0, 0, 0};
        iv8 z8 = {0, 0, 0, 0, 0, 0, 0, 0};
        // this toolchain exposes the 6-arg (clang-23 / therock) signature
        __builtin_amdgcn_tensor_load_to_lds(g0, g1, z4, z4, z8, 0);
        __builtin_amdgcn_s_wait_tensorcnt(0);
    }
#else
    for (int i = threadIdx.x * 4; i < WWORDS; i += 128 * 4)
        *(uint4*)&wlds[i] = *(const uint4*)&wp[i];
#endif
    __syncthreads();

    const int tiles  = (E + 15) >> 4;
    const int groups = (tiles + 3) >> 2;

    for (int grp = blockIdx.x; grp < groups; grp += gridDim.x) {
        const int e0 = (grp * 4 + wave) * 16;

        // ---- per-edge setup: indices + sh = sqrt3 * r_hat ----
        if (lane < 16) {
            int eg = e0 + lane;
            int si = 0, di = 0;
            float shx = 0.f, shy = 0.f, shz = 0.f;
            if (eg < E) {
                si = eidx[eg];
                di = eidx[E + eg];
                float vx = evec[3 * eg + 0], vy = evec[3 * eg + 1], vz = evec[3 * eg + 2];
                float inv = SQ3 / (sqrtf(vx * vx + vy * vy + vz * vz) + 1e-12f);
                shx = vx * inv; shy = vy * inv; shz = vz * inv;
            }
            L.srcv[lane] = si; L.dstv[lane] = di;
            L.sh[lane][0] = shx; L.sh[lane][1] = shy; L.sh[lane][2] = shz;
        }
        __syncthreads();

        // ---- gather + de-interleave (x,y,z per channel) into bf16 stream tiles ----
        for (int e = 0; e < 16; ++e) {
            const float* row = atom + (size_t)L.srcv[e] * 256;
            float4 p0 = *(const float4*)(row + lane * 8);
            float4 p1 = *(const float4*)(row + lane * 8 + 4);
            float vals[8] = {p0.x, p0.y, p0.z, p0.w, p1.x, p1.y, p1.z, p1.w};
#pragma unroll
            for (int j = 0; j < 8; ++j) {
                int ch = lane * 8 + j;
                if (ch < 64) {
                    L.xs[0][e * 64 + ch] = f2bf(vals[j]);
                } else {
                    int cc = ch - 64; int u = cc / 3; int m3 = cc - 3 * u;
                    L.xs[1 + m3][e * 64 + u] = f2bf(vals[j]);
                }
            }
        }
        __syncthreads();

        v8f c[4], c01[4];

        // ---- lin_src: s0 = x0@ls0*INV_MUL ; s1m = x1m@ls1*INV_MUL ----
        for (int st = 0; st < 4; ++st) {
            mm1(c, L.xs[st], wlds, st == 0 ? SLOT_LS0 : SLOT_LS1, lane);
#pragma unroll
            for (int nb = 0; nb < 4; ++nb)
#pragma unroll
                for (int v = 0; v < 8; ++v) {
                    int mr = v + 8 * g, n = nb * 16 + ln;
                    L.s[st][mr * 64 + n] = f2bf(c[nb][v] * INV_MUL);
                }
        }
        __syncthreads();

        // ---- fctp1: dot = <s1, sh>/sqrt3 ----
        for (int i = 0; i < 32; ++i) {
            int t = i * 32 + lane, e = t >> 6, u = t & 63;
            float dv = (bf2f(L.s[1][e * 64 + u]) * L.sh[e][0]
                      + bf2f(L.s[2][e * 64 + u]) * L.sh[e][1]
                      + bf2f(L.s[3][e * 64 + u]) * L.sh[e][2]) * INV_SQ3;
            L.dott[t] = f2bf(dv);
        }
        __syncthreads();

        // h0 = (s0@tp1_w00 + dot@tp1_w11) * INV_TP  -> xs[0]
        mm2(c, L.s[0], SLOT_T1W00, L.dott, SLOT_T1W11, wlds, lane);
#pragma unroll
        for (int nb = 0; nb < 4; ++nb)
#pragma unroll
            for (int v = 0; v < 8; ++v) {
                int mr = v + 8 * g, n = nb * 16 + ln;
                L.xs[0][mr * 64 + n] = f2bf(c[nb][v] * INV_TP);
            }
        // t01 = s0@tp1_w01 (register resident; same C-layout as h1 accumulators)
        mm1(c01, L.s[0], wlds, SLOT_T1W01, lane);
        // h1m = (t01*sh_m + s1m@tp1_w10) * INV_TP -> xs[1+m]
        for (int mc = 0; mc < 3; ++mc) {
            mm1(c, L.s[1 + mc], wlds, SLOT_T1W10, lane);
#pragma unroll
            for (int nb = 0; nb < 4; ++nb)
#pragma unroll
                for (int v = 0; v < 8; ++v) {
                    int mr = v + 8 * g, n = nb * 16 + ln;
                    float val = (c[nb][v] + c01[nb][v] * L.sh[mr][mc]) * INV_TP;
                    L.xs[1 + mc][mr * 64 + n] = f2bf(val);
                }
        }
        __syncthreads();

        // ---- lin_hidden: h0' (+ leaky_relu tile) ; h1m' ----
        mm1(c, L.xs[0], wlds, SLOT_LH0, lane);
#pragma unroll
        for (int nb = 0; nb < 4; ++nb)
#pragma unroll
            for (int v = 0; v < 8; ++v) {
                int mr = v + 8 * g, n = nb * 16 + ln;
                float val = c[nb][v] * INV_MUL;
                L.s[0][mr * 64 + n] = f2bf(val);
                L.h0r[mr * 64 + n]  = f2bf(val > 0.f ? val : 0.01f * val);
            }
        for (int mc = 0; mc < 3; ++mc) {
            mm1(c, L.xs[1 + mc], wlds, SLOT_LH1, lane);
#pragma unroll
            for (int nb = 0; nb < 4; ++nb)
#pragma unroll
                for (int v = 0; v < 8; ++v) {
                    int mr = v + 8 * g, n = nb * 16 + ln;
                    L.s[1 + mc][mr * 64 + n] = f2bf(c[nb][v] * INV_MUL);
                }
        }
        __syncthreads();

        // ---- sc = leaky_relu(h0') @ lsc * INV_MUL ----
        mm1(c, L.h0r, wlds, SLOT_LSC, lane);
#pragma unroll
        for (int nb = 0; nb < 4; ++nb)
#pragma unroll
            for (int v = 0; v < 8; ++v) {
                int mr = v + 8 * g, n = nb * 16 + ln;
                L.act[mr * 64 + n] = c[nb][v] * INV_MUL;
            }
        __syncthreads();

        // ---- softmax over [sc(64), zeros(192)] ----
        if (lane < 16) {
            float mx = 0.f;                       // max includes the 192 zero channels
            for (int i = 0; i < 64; ++i) mx = fmaxf(mx, L.act[lane * 64 + i]);
            float ssum = 0.f;
            for (int i = 0; i < 64; ++i) {
                float ev = __expf(L.act[lane * 64 + i] - mx);
                L.act[lane * 64 + i] = ev; ssum += ev;
            }
            float z = __expf(-mx);
            float invZ = 1.f / (ssum + 192.f * z);
            for (int i = 0; i < 64; ++i) L.act[lane * 64 + i] *= invZ;
            L.a1[lane] = z * invZ;
        }
        __syncthreads();

        // ---- fctp2: dot2 = <h1', sh>/sqrt3 ----
        for (int i = 0; i < 32; ++i) {
            int t = i * 32 + lane, e = t >> 6, u = t & 63;
            float dv = (bf2f(L.s[1][e * 64 + u]) * L.sh[e][0]
                      + bf2f(L.s[2][e * 64 + u]) * L.sh[e][1]
                      + bf2f(L.s[3][e * 64 + u]) * L.sh[e][2]) * INV_SQ3;
            L.dott[t] = f2bf(dv);
        }
        __syncthreads();

        // o0 = (h0'@tp2_w00 + dot2@tp2_w11)*INV_TP ; msg0 = act*o0 -> scatter-add
        mm2(c, L.s[0], SLOT_T2W00, L.dott, SLOT_T2W11, wlds, lane);
#pragma unroll
        for (int nb = 0; nb < 4; ++nb)
#pragma unroll
            for (int v = 0; v < 8; ++v) {
                int mr = v + 8 * g, n = nb * 16 + ln;
                if (e0 + mr < E) {
                    float val = c[nb][v] * INV_TP * L.act[mr * 64 + n];
                    atomic_add_f32(out + (size_t)L.dstv[mr] * 256 + n, val);
                }
            }
        // t01b = h0'@tp2_w01 (registers)
        mm1(c01, L.s[0], wlds, SLOT_T2W01, lane);
        // o1m = (t01b*sh_m + h1m'@tp2_w10)*INV_TP ; msg1 = a1*o1m -> scatter (interleaved)
        for (int mc = 0; mc < 3; ++mc) {
            mm1(c, L.s[1 + mc], wlds, SLOT_T2W10, lane);
#pragma unroll
            for (int nb = 0; nb < 4; ++nb)
#pragma unroll
                for (int v = 0; v < 8; ++v) {
                    int mr = v + 8 * g, n = nb * 16 + ln;
                    if (e0 + mr < E) {
                        float val = (c[nb][v] + c01[nb][v] * L.sh[mr][mc]) * INV_TP * L.a1[mr];
                        atomic_add_f32(out + (size_t)L.dstv[mr] * 256 + 64 + n * 3 + mc, val);
                    }
                }
        }
        __syncthreads();   // protect srcv/dstv/sh before next iteration's setup
    }
}

extern "C" void kernel_launch(void* const* d_in, const int* in_sizes, int n_in,
                              void* d_out, int out_size, void* d_ws, size_t ws_size,
                              hipStream_t stream) {
    const float* W[15];
    for (int i = 0; i < 15; ++i) W[i] = (const float*)d_in[i];
    const float* atom = (const float*)d_in[15];
    const float* evec = (const float*)d_in[16];
    const int*   eidx = (const int*)d_in[17];
    const int N = in_sizes[15] / 256;
    const int E = in_sizes[17] / 2;
    uint32_t* wp = (uint32_t*)d_ws;
    float* out = (float*)d_out;

    pack_weights_kernel<<<(WWORDS + 255) / 256, 256, 0, stream>>>(
        W[0], W[1], W[2], W[3], W[4], W[5], W[6], W[7],
        W[8], W[9], W[10], W[11], W[12], W[13], W[14], wp);

    int total = N * 256;
    init_out_kernel<<<(total + 255) / 256, 256, 0, stream>>>(atom, out, total);

    int tiles  = (E + 15) / 16;
    int groups = (tiles + 3) / 4;
    int grid   = groups < 512 ? groups : 512;   // persistent blocks; weights staged once each
    equiformer_edge_kernel<<<grid, 128, 0, stream>>>(atom, evec, eidx, wp, out, E);
}
